// HMTCL_1872605741068
// MI455X (gfx1250) — matrix operand order repaired and checked by reading, not compiled, not run on hardware
//
#include <hip/hip_runtime.h>
#include <hip/hip_bf16.h>
#include <math.h>

typedef __attribute__((ext_vector_type(16))) _Float16 v16h;
typedef __attribute__((ext_vector_type(8)))  _Float16 v8h;
typedef __attribute__((ext_vector_type(8)))  float    v8f;

#define CDIV(a,b) (((a)+(b)-1)/(b))
#define LDS_STRIDE 136   // 64 rows x 136 halves; row stride 272B avoids ds bank conflicts

// fragment assembly from two 16-byte (8-half) loads
__device__ inline v16h frag2(const _Float16* p0, const _Float16* p1){
  v8h x = *(const v8h*)p0;
  v8h y = *(const v8h*)p1;
  return __builtin_shufflevector(x, y, 0,1,2,3,4,5,6,7,8,9,10,11,12,13,14,15);
}

// ---------------- fills ----------------
__global__ void k_fill_f32(float* p, float v, int n){
  int i = blockIdx.x*blockDim.x + threadIdx.x;
  if (i < n) p[i] = v;
}
__global__ void k_fill_u32(unsigned* p, unsigned v, int n){
  int i = blockIdx.x*blockDim.x + threadIdx.x;
  if (i < n) p[i] = v;
}

// ---------------- f16 repack kernels ----------------
__global__ void k_cvt16(const float* __restrict__ src, _Float16* dst, int n){
  int i = blockIdx.x*blockDim.x + threadIdx.x;
  if (i < n) dst[i] = (_Float16)src[i];
}
// transpose: src (R x C) f32 -> dst (C x R) f16
__global__ void k_cvt16_t(const float* __restrict__ src, _Float16* dst, int R, int Cc){
  int idx = blockIdx.x*blockDim.x + threadIdx.x;
  if (idx >= R*Cc) return;
  int c = idx / R, r = idx - c*R;
  dst[idx] = (_Float16)src[(size_t)r*Cc + c];
}
// pad cols: src (Rs x Cs) f32 -> dst (Rs x Cp) f16, zeros beyond Cs
__global__ void k_pad_a16(const float* __restrict__ src, _Float16* dst, int Rs, int Cs, int Cp){
  int idx = blockIdx.x*blockDim.x + threadIdx.x;
  if (idx >= Rs*Cp) return;
  int i = idx / Cp, j = idx - i*Cp;
  dst[idx] = (j < Cs) ? (_Float16)src[(size_t)i*Cs + j] : (_Float16)0.f;
}
// transpose+pad rows: src (Rs x C) f32 -> dst (C x Rp) f16, zeros for r >= Rs
__global__ void k_padT_b16(const float* __restrict__ src, _Float16* dst, int Rs, int Rp, int Cc){
  int idx = blockIdx.x*blockDim.x + threadIdx.x;
  if (idx >= Cc*Rp) return;
  int c = idx / Rp, r = idx - c*Rp;
  dst[idx] = (r < Rs) ? (_Float16)src[(size_t)r*Cc + c] : (_Float16)0.f;
}
// conv weights (O,C,KW) f32 -> dst [k][o][c] f16
__global__ void k_pack_convw(const float* __restrict__ w, _Float16* dst, int O, int Cc, int KW){
  int idx = blockIdx.x*blockDim.x + threadIdx.x;
  if (idx >= KW*O*Cc) return;
  int k = idx / (O*Cc);
  int rem = idx - k*O*Cc;
  int o = rem / Cc, i = rem - o*Cc;
  dst[idx] = (_Float16)w[((size_t)o*Cc + i)*KW + k];
}

// ---------------- LDS-staged WMMA GEMM: C = act(A@B + bias) ----------------
// 4-wave block computes a 64x64 tile. A: MxK f16 (lda). BT: NxK f16 (ldbt). C: MxN f32.
// K must be a multiple of 128. The shared 64xK B panel is staged through LDS in
// 64x128 superchunks; B fragments come from ds_load_b128, A from global b128.
__global__ __launch_bounds__(128)
void k_wmma_gemm16_lds(const _Float16* __restrict__ A, const _Float16* __restrict__ BT,
                       float* __restrict__ C, int K, int lda, int ldbt, int ldc,
                       const float* __restrict__ bias, int act)
{
  __shared__ _Float16 BS[64*LDS_STRIDE];
  const int tid  = threadIdx.x;
  const int w    = tid >> 5;
  const int lane = tid & 31;
  const int g = lane >> 4;
  const int r = lane & 15;
  const int tm = blockIdx.y*64 + w*16;
  const int tn = blockIdx.x*64;
  const _Float16* Ar = A + (size_t)(tm + r) * lda + 8*g;

  v8f acc[4];
#pragma unroll
  for (int t = 0; t < 4; ++t) acc[t] = (v8f){};

  for (int ks = 0; ks < K; ks += 128) {
    __syncthreads();
#pragma unroll
    for (int u = tid; u < 1024; u += 128) {           // 64 rows x 16 v8h units
      int row = u >> 4, c8 = (u & 15) * 8;
      *(v8h*)(BS + row*LDS_STRIDE + c8) =
          *(const v8h*)(BT + (size_t)(tn + row)*ldbt + ks + c8);
    }
    __syncthreads();
#pragma unroll
    for (int kc = 0; kc < 128; kc += 32) {
      v16h a = frag2(Ar + ks + kc, Ar + ks + kc + 16);
#pragma unroll
      for (int t = 0; t < 4; ++t) {
        const _Float16* bp = BS + (t*16 + r)*LDS_STRIDE + kc + 8*g;
        v16h b = frag2(bp, bp + 16);
        acc[t] = __builtin_amdgcn_wmma_f32_16x16x32_f16(false, a, false, b, (short)0, acc[t], false, false);
      }
    }
  }
#pragma unroll
  for (int t = 0; t < 4; ++t) {
    int cn = tn + t*16 + r;
    float bv = bias ? bias[cn] : 0.f;
#pragma unroll
    for (int r2 = 0; r2 < 8; ++r2) {
      int row = tm + r2 + 8*g;
      float v = acc[t][r2] + bv;
      if (act == 1)      v = fmaxf(v, 0.f);
      else if (act == 2) v = (v > 0.f) ? v : (expf(v) - 1.f);
      C[(size_t)row * ldc + cn] = v;
    }
  }
}

// ---------------- small direct GEMM (N=32, row-gather) for the MLP head ----------------
__global__ __launch_bounds__(32)
void k_wmma_gemm16_sm(const _Float16* __restrict__ A, const _Float16* __restrict__ BT,
                      float* __restrict__ C, int K, int lda, int ldbt, int ldc,
                      int act, const int* __restrict__ rowIdx)
{
  const int lane = threadIdx.x;
  const int g = lane >> 4;
  const int r = lane & 15;
  const int tm = blockIdx.y * 16;
  const int tn = blockIdx.x * 32;

  int am = tm + r;
  if (rowIdx) am = rowIdx[am];
  const _Float16* Ar = A + (size_t)am * lda;

  const _Float16* Br[2];
#pragma unroll
  for (int t = 0; t < 2; ++t) Br[t] = BT + (size_t)(tn + t*16 + r) * ldbt + 8*g;

  v8f acc[2];
#pragma unroll
  for (int t = 0; t < 2; ++t) acc[t] = (v8f){};

  for (int kc = 0; kc < K; kc += 32) {
    v16h a = frag2(Ar + kc + 8*g, Ar + kc + 16 + 8*g);
#pragma unroll
    for (int t = 0; t < 2; ++t) {
      v16h b = frag2(Br[t] + kc, Br[t] + kc + 16);
      acc[t] = __builtin_amdgcn_wmma_f32_16x16x32_f16(false, a, false, b, (short)0, acc[t], false, false);
    }
  }
#pragma unroll
  for (int t = 0; t < 2; ++t) {
#pragma unroll
    for (int r2 = 0; r2 < 8; ++r2) {
      int row = tm + r2 + 8*g;
      float v = acc[t][r2];
      if (act == 2) v = (v > 0.f) ? v : (expf(v) - 1.f);
      C[(size_t)row * ldc + tn + t*16 + r] = v;
    }
  }
}

// ---------------- LDS-staged WMMA conv1d on (B, L, C) layout + bias + relu --------------
// y[b][l][o] = relu( sum_k sum_i XT[b][l+k][i] * W16[k][o][i] + bias[o] ), C=O=128.
// 4-wave block: 64 output rows x 64 channels; per-tap 64x128 weight panel staged in LDS.
template<int KW>
__global__ __launch_bounds__(128)
void k_wmma_conv16(const _Float16* __restrict__ XT, const _Float16* __restrict__ W16,
                   const float* __restrict__ bias, float* __restrict__ Y,
                   int Lin, int Lout)
{
  __shared__ _Float16 BS[64*LDS_STRIDE];
  const int tid  = threadIdx.x;
  const int w    = tid >> 5;
  const int lane = tid & 31;
  const int g = lane >> 4;
  const int r = lane & 15;
  const int tm = blockIdx.y*64 + w*16;  // rows over B*Lout
  const int tn = blockIdx.x*64;         // output channels

  int m = tm + r;
  int bb = m / Lout;
  int ll = m - bb * Lout;
  const _Float16* Ab = XT + (size_t)(bb*Lin + ll) * 128 + 8*g;

  v8f acc[4];
#pragma unroll
  for (int t = 0; t < 4; ++t) acc[t] = (v8f){};

  for (int k = 0; k < KW; ++k) {
    __syncthreads();
#pragma unroll
    for (int u = tid; u < 1024; u += 128) {
      int row = u >> 4, c8 = (u & 15) * 8;
      *(v8h*)(BS + row*LDS_STRIDE + c8) =
          *(const v8h*)(W16 + ((size_t)k*128 + tn + row)*128 + c8);
    }
    __syncthreads();
    const _Float16* Ar = Ab + (size_t)k * 128;
#pragma unroll
    for (int kc = 0; kc < 128; kc += 32) {
      v16h a = frag2(Ar + kc, Ar + kc + 16);
#pragma unroll
      for (int t = 0; t < 4; ++t) {
        const _Float16* bp = BS + (t*16 + r)*LDS_STRIDE + kc + 8*g;
        v16h b = frag2(bp, bp + 16);
        acc[t] = __builtin_amdgcn_wmma_f32_16x16x32_f16(false, a, false, b, (short)0, acc[t], false, false);
      }
    }
  }
#pragma unroll
  for (int t = 0; t < 4; ++t) {
    int o = tn + t*16 + r;
    float bv = bias[o];
#pragma unroll
    for (int r2 = 0; r2 < 8; ++r2) {
      int row = tm + r2 + 8*g;
      int b2 = row / Lout, l2 = row - b2*Lout;
      Y[((size_t)b2*Lout + l2)*128 + o] = fmaxf(acc[t][r2] + bv, 0.f);
    }
  }
}

// ---------------- graph conv helpers ----------------
__global__ void k_deg(const int* src, const int* dst, int E, float* degS, float* degD){
  int e = blockIdx.x*blockDim.x + threadIdx.x;
  if (e < E){ atomicAdd(&degS[src[e]], 1.f); atomicAdd(&degD[dst[e]], 1.f); }
}
__global__ void k_scatter(const float* __restrict__ H, const int* src, const int* dst,
                          const float* degS, int E, int F, float* Msg){
  int idx = blockIdx.x*blockDim.x + threadIdx.x;
  if (idx >= E*F) return;
  int e = idx / F, f = idx - e*F;
  int s = src[e];
  float ds = degS[s];
  float coef = (ds > 0.f) ? rsqrtf(fmaxf(ds, 1.f)) : 0.f;
  atomicAdd(&Msg[(size_t)dst[e]*F + f], H[(size_t)s*F + f]*coef);
}
__global__ void k_scale_rows(float* Msg, const float* degD, int Nn, int F){
  int idx = blockIdx.x*blockDim.x + threadIdx.x;
  if (idx >= Nn*F) return;
  int n = idx / F;
  float dd = degD[n];
  float coef = (dd > 0.f) ? rsqrtf(fmaxf(dd, 1.f)) : 0.f;
  Msg[idx] *= coef;
}

// ---------------- semantic attention ----------------
__global__ void k_sematt_w(const float* __restrict__ Z, long sB, long sM, long sD, int Mn,
                           const float* __restrict__ W1, const float* __restrict__ b1,
                           const float* __restrict__ W2, float* wsum){
  int bm = blockIdx.x;
  const float* zr = Z + (long)(bm / Mn)*sB + (long)(bm % Mn)*sM;
  int j = threadIdx.x;                 // 32 threads
  float acc = b1[j];
  for (int d = 0; d < 128; ++d) acc += zr[(long)d*sD] * W1[d*32 + j];
  float t = tanhf(acc) * W2[j];
  for (int off = 16; off > 0; off >>= 1) t += __shfl_xor(t, off, 32);
  if (j == 0) atomicAdd(&wsum[bm % Mn], t);
}
__global__ void k_sematt_beta(const float* wsum, float* beta, int Mn, int Bn){
  if (threadIdx.x == 0 && blockIdx.x == 0){
    float mx = -1e30f;
    for (int m = 0; m < Mn; ++m) mx = fmaxf(mx, wsum[m] / Bn);
    float s = 0.f;
    for (int m = 0; m < Mn; ++m){ float e = expf(wsum[m]/Bn - mx); beta[m] = e; s += e; }
    for (int m = 0; m < Mn; ++m) beta[m] /= s;
  }
}
__global__ void k_sematt_combine(const float* __restrict__ Z, long sB, long sM, long sD,
                                 int Bn, int Mn, const float* beta, float* Out, int ldo){
  int idx = blockIdx.x*blockDim.x + threadIdx.x;
  if (idx >= Bn*128) return;
  int b = idx >> 7, dd = idx & 127;
  float s = 0.f;
  for (int m = 0; m < Mn; ++m) s += beta[m] * Z[(long)b*sB + (long)m*sM + (long)dd*sD];
  Out[(size_t)b*ldo + dd] = s;
}

// ---------------- ProteinCNN: embed (to (B,L,C)) + BN on channel-inner layout ------------
__global__ void k_embed(const float* __restrict__ emb, const int* __restrict__ vp, float* X){
  int idx = blockIdx.x*blockDim.x + threadIdx.x;     // (B*L) x C
  if (idx >= 128*512*128) return;
  int row = idx >> 7, c = idx & 127;
  X[idx] = emb[(size_t)vp[row]*128 + c];
}
__global__ void k_bn_stats(const float* __restrict__ X, int Rows, float* mv){
  int o = blockIdx.x, tid = threadIdx.x;            // 128 channels, channel-inner
  float s = 0.f, s2 = 0.f;
  for (int i = tid; i < Rows; i += blockDim.x){
    float v = X[(size_t)i*128 + o];
    s += v; s2 += v*v;
  }
  __shared__ float sh[256], sh2[256];
  sh[tid] = s; sh2[tid] = s2; __syncthreads();
  for (int st = 128; st > 0; st >>= 1){
    if (tid < st){ sh[tid] += sh[tid+st]; sh2[tid] += sh2[tid+st]; }
    __syncthreads();
  }
  if (tid == 0){ float m = sh[0]/Rows; mv[o] = m; mv[128+o] = sh2[0]/Rows - m*m; }
}
__global__ void k_bn_apply(float* X, const float* mv, const float* gamma, const float* beta, int n){
  int idx = blockIdx.x*blockDim.x + threadIdx.x;
  if (idx >= n) return;
  int o = idx & 127;
  float v = X[idx];
  X[idx] = (v - mv[o]) * rsqrtf(mv[128+o] + 1e-5f) * gamma[o] + beta[o];
}

// ---------------- pair features (built directly in f16) ----------------
__global__ void k_build_feat16(const float* __restrict__ dmat, const float* __restrict__ pmat,
                               const int* __restrict__ pairs, _Float16* feat){
  int idx = blockIdx.x*blockDim.x + threadIdx.x;     // 32768 x 512
  if (idx >= 32768*512) return;
  int i = idx >> 9, c = idx & 511;
  float v = (c < 256) ? dmat[(size_t)pairs[2*i]*256 + c]
                      : pmat[(size_t)pairs[2*i+1]*256 + (c-256)];
  feat[idx] = (_Float16)((v > 0.f) ? 1.f : v);
}

// ---------------- GAT ----------------
__device__ inline unsigned fenc(float x){
  unsigned u = __float_as_uint(x);
  return (u & 0x80000000u) ? ~u : (u | 0x80000000u);
}
__device__ inline float fdec(unsigned u){
  unsigned b = (u & 0x80000000u) ? (u & 0x7FFFFFFFu) : ~u;
  return __uint_as_float(b);
}
__global__ void k_gat_elr(const float* __restrict__ Fm, const float* al, const float* ar,
                          float* el, float* er){
  int idx = blockIdx.x*blockDim.x + threadIdx.x;     // 32768 * 4
  if (idx >= 32768*4) return;
  int n = idx >> 2, h = idx & 3;
  float a = 0.f, b = 0.f;
  for (int c = 0; c < 32; ++c){
    float f = Fm[(size_t)n*128 + h*32 + c];
    a += f * al[h*32 + c];
    b += f * ar[h*32 + c];
  }
  el[idx] = a; er[idx] = b;
}
__global__ void k_gat_logits(const float* el, const float* er, const int* ks, const int* kd,
                             int Eknn, int Etot, float* ebuf, unsigned* mmax){
  int idx = blockIdx.x*blockDim.x + threadIdx.x;
  if (idx >= Etot*4) return;
  int e = idx >> 2, h = idx & 3;
  int s, t;
  if (e < Eknn){ s = ks[e]; t = kd[e]; } else { s = t = e - Eknn; }
  float v = el[s*4 + h] + er[t*4 + h];
  v = (v >= 0.f) ? v : 0.2f * v;                      // leaky_relu 0.2
  ebuf[idx] = v;
  atomicMax(&mmax[t*4 + h], fenc(v));
}
__global__ void k_gat_exp(float* ebuf, const unsigned* mmax, const int* ks, const int* kd,
                          int Eknn, int Etot, float* ssum){
  int idx = blockIdx.x*blockDim.x + threadIdx.x;
  if (idx >= Etot*4) return;
  int e = idx >> 2, h = idx & 3;
  int t = (e < Eknn) ? kd[e] : (e - Eknn);
  float ex = expf(ebuf[idx] - fdec(mmax[t*4 + h]));
  ebuf[idx] = ex;
  atomicAdd(&ssum[t*4 + h], ex);
}
__global__ void k_gat_bias(float* Out, const float* gb){
  int idx = blockIdx.x*blockDim.x + threadIdx.x;     // 32768 * 128
  if (idx >= 32768*128) return;
  Out[idx] = gb[idx & 127];
}
__global__ void k_gat_agg(const float* __restrict__ ebuf, const float* __restrict__ ssum,
                          const float* __restrict__ Fm, const int* ks, const int* kd,
                          int Eknn, int Etot, float* Out){
  int idx = blockIdx.x*blockDim.x + threadIdx.x;     // Etot * 128
  if (idx >= Etot*128) return;
  int e = idx >> 7, c = idx & 127, h = c >> 5;
  int s, t;
  if (e < Eknn){ s = ks[e]; t = kd[e]; } else { s = t = e - Eknn; }
  float a = ebuf[e*4 + h] / (ssum[t*4 + h] + 1e-9f);
  atomicAdd(&Out[(size_t)t*128 + c], a * Fm[(size_t)s*128 + c]);
}
__global__ void k_elu(float* p, int n){
  int i = blockIdx.x*blockDim.x + threadIdx.x;
  if (i < n){ float v = p[i]; p[i] = (v > 0.f) ? v : (expf(v) - 1.f); }
}

// ---------------- final MLP tail + log_softmax ----------------
__global__ void k_mlp2(const float* __restrict__ Hd, const float* __restrict__ W2,
                       float* out, int Nr){
  int i = blockIdx.x*blockDim.x + threadIdx.x;
  if (i >= Nr) return;
  float l0 = 0.f, l1 = 0.f;
  for (int j = 0; j < 32; ++j){
    float h = Hd[i*32 + j];
    l0 += h * W2[2*j];
    l1 += h * W2[2*j + 1];
  }
  float mx = fmaxf(l0, l1);
  float lse = mx + logf(expf(l0 - mx) + expf(l1 - mx));
  out[2*i]     = l0 - lse;
  out[2*i + 1] = l1 - lse;
}

// =======================================================================
extern "C" void kernel_launch(void* const* d_in, const int* in_sizes, int n_in,
                              void* d_out, int out_size, void* d_ws, size_t ws_size,
                              hipStream_t stream) {
  (void)in_sizes; (void)n_in; (void)out_size; (void)ws_size;
  float* ws  = (float*)d_ws;
  float* out = (float*)d_out;

  // ----- workspace layout (floats) -----
  const size_t BUF0 = 0;                               // 8,388,608  XT0/XT2 f32; later feat f16
  const size_t BUF1 = BUF0 + (size_t)128*128*512;      // 8,388,608  XT1/XT3 f32; later hcat f16
  const size_t FBUF = BUF1 + (size_t)128*128*512;      // 4,194,304  hm ping-pong; CNN A16; GAT f
  const size_t OBUF = FBUF + (size_t)32768*128;        // 4,194,304  GAT out f32 (hcat)
  const size_t MSG  = OBUF + (size_t)32768*128;        //   786,432  graph-conv messages
  const size_t EBUF = MSG  + (size_t)6144*128;         //   786,432  edge logits/ex
  const size_t ELR  = EBUF + (size_t)196608*4;         //   262,144  el | er
  const size_t MAXS = ELR  + (size_t)32768*8;          //   131,072  seg-max (encoded)
  const size_t SUMS = MAXS + (size_t)32768*4;          //   131,072  seg-sum
  const size_t ZBUF = SUMS + (size_t)32768*4;          //    32,768  zd/zp (128,2,128)
  const size_t TMPD = ZBUF + 32768;                    //    16,384  sem-att output
  const size_t HID  = TMPD + 16384;                    //   262,144  MLP hidden
  const size_t WSEM = HID  + (size_t)8192*32;          //     1,024  wsum | beta
  const size_t DEG  = WSEM + 1024;                     //    12,288  degS | degD
  const size_t MEANV= DEG  + 12288;                    //       256  BN mean|var
  const size_t A16S = MEANV + 256;                     //   524,288  f16 A staging (1M halves)
  const size_t B16S = A16S + 524288;                   //    81,920  f16 BT / convW staging

  _Float16* a16 = (_Float16*)(ws + A16S);
  _Float16* b16 = (_Float16*)(ws + B16S);

  const int Eknn = 163840, Etot = 196608;

  auto gemm16 = [&](const _Float16* A, const _Float16* BT, float* C, int M, int N, int K,
                    int lda, int ldbt, int ldc, const float* bias, int act){
    k_wmma_gemm16_lds<<<dim3(N/64, M/64), 128, 0, stream>>>(A,BT,C,K,lda,ldbt,ldc,bias,act);
  };
  // graph conv: BT16 (128 x F) must already be staged in b16
  auto graph_conv = [&](const float* H, const int* src, const int* dst, int E,
                        int n, int F, const float* bias, float* Cout, int ldc){
    float* degS = ws + DEG; float* degD = ws + DEG + 6144;
    k_fill_f32<<<CDIV(12288,256),256,0,stream>>>(degS, 0.f, 12288);
    k_deg<<<CDIV(E,256),256,0,stream>>>(src,dst,E,degS,degD);
    k_fill_f32<<<CDIV(n*F,256),256,0,stream>>>(ws+MSG, 0.f, n*F);
    k_scatter<<<CDIV(E*F,256),256,0,stream>>>(H,src,dst,degS,E,F,ws+MSG);
    k_scale_rows<<<CDIV(n*F,256),256,0,stream>>>(ws+MSG,degD,n,F);
    k_cvt16<<<CDIV(n*F,256),256,0,stream>>>(ws+MSG, a16, n*F);
    gemm16(a16, b16, Cout, n, 128, F, F, F, ldc, bias, /*relu*/1);
  };
  auto sem_att = [&](const float* Z, long sB, long sM, long sD, int Bn, int Mn,
                     const float* W1, const float* b1, const float* W2,
                     float* Out, int ldo){
    float* wsum = ws + WSEM; float* beta = ws + WSEM + 512;
    k_fill_f32<<<CDIV(Mn,256),256,0,stream>>>(wsum, 0.f, Mn);
    k_sematt_w<<<Bn*Mn, 32, 0, stream>>>(Z, sB,sM,sD, Mn, W1, b1, W2, wsum);
    k_sematt_beta<<<1,1,0,stream>>>(wsum, beta, Mn, Bn);
    k_sematt_combine<<<CDIV(Bn*128,256),256,0,stream>>>(Z,sB,sM,sD,Bn,Mn,beta,Out,ldo);
  };

  // ---------------- HAN_DTI (drug, protein) ----------------
  for (int side = 0; side < 2; ++side){
    const float* h    = (const float*)d_in[side ? 1 : 0];
    const int*   mp   = (const int*)  d_in[side ? 39 : 38];
    const float* Wh   = (const float*)d_in[side ? 8 : 2];
    const float* bh   = (const float*)d_in[side ? 9 : 3];
    const float* sW1  = (const float*)d_in[side ? 10 : 4];
    const float* sb1  = (const float*)d_in[side ? 11 : 5];
    const float* sW2  = (const float*)d_in[side ? 12 : 6];
    const float* pred = (const float*)d_in[side ? 13 : 7];
    float* dsec = out + (side ? 49152 : 16384);
    k_cvt16_t<<<CDIV(256*128,256),256,0,stream>>>(Wh, b16, 256, 128);     // BT 128x256
    for (int i = 0; i < 2; ++i)
      graph_conv(h, mp + i*4096, mp + i*4096 + 2048, 2048, 128, 256,
                 bh, ws+ZBUF + i*128, 256);
    sem_att(ws+ZBUF, 256, 128, 1, 128, 2, sW1, sb1, sW2, ws+TMPD, 128);
    k_cvt16<<<CDIV(16384,256),256,0,stream>>>(ws+TMPD, a16, 16384);
    k_cvt16_t<<<CDIV(128*128,256),256,0,stream>>>(pred, b16, 128, 128);   // BT 128x128
    gemm16(a16, b16, dsec, 128,128,128, 128,128,256, nullptr, 0);
  }

  // ---------------- MolecularGCN ----------------
  // pad atom_feats (6144x75 -> 6144x128 f16) and init_W (75x128 -> BT 128x128 f16), K=128
  k_pad_a16<<<CDIV(6144*128,256),256,0,stream>>>((const float*)d_in[14], a16, 6144, 75, 128);
  k_padT_b16<<<CDIV(128*128,256),256,0,stream>>>((const float*)d_in[15], b16, 75, 128, 128);
  float* hmA = ws + FBUF;
  float* hmB = hmA + (size_t)6144*128;
  gemm16(a16, b16, hmA, 6144,128,128, 128,128,128, nullptr, 0);
  const int* mol = (const int*)d_in[40];
  const float* gW = (const float*)d_in[16];
  const float* gB = (const float*)d_in[17];
  for (int l = 0; l < 3; ++l){
    k_cvt16_t<<<CDIV(128*128,256),256,0,stream>>>(gW + (size_t)l*128*128, b16, 128, 128);
    graph_conv(hmA, mol, mol+24576, 24576, 6144, 128, gB + l*128, hmB, 128);
    float* t = hmA; hmA = hmB; hmB = t;
  }
  sem_att(hmA, 48*128, 128, 1, 128, 48,
          (const float*)d_in[18], (const float*)d_in[19], (const float*)d_in[20],
          out + 16384 + 128, 256);

  // ---------------- ProteinCNN (activations kept in (B, L, C) layout) ----------------
  _Float16* xt16 = (_Float16*)(ws + FBUF);   // CNN f16 staging (8,388,608 halves max)
  k_embed<<<CDIV(128*512*128,256),256,0,stream>>>((const float*)d_in[21], (const int*)d_in[36], ws+BUF0);
  // conv1 (KW=3): BUF0 (B,512,128) -> BUF1 (B,510,128)
  k_cvt16<<<CDIV(128*512*128,256),256,0,stream>>>(ws+BUF0, xt16, 128*512*128);
  k_pack_convw<<<CDIV(3*128*128,256),256,0,stream>>>((const float*)d_in[22], b16, 128, 128, 3);
  k_wmma_conv16<3><<<dim3(2,(128*510)/64),128,0,stream>>>(xt16, b16, (const float*)d_in[23],
      ws+BUF1, 512, 510);
  k_bn_stats<<<128,256,0,stream>>>(ws+BUF1, 128*510, ws+MEANV);
  k_bn_apply<<<CDIV(128*510*128,256),256,0,stream>>>(ws+BUF1, ws+MEANV,
      (const float*)d_in[28] + 0, (const float*)d_in[29] + 0, 128*510*128);
  // conv2 (KW=6): BUF1 -> BUF0
  k_cvt16<<<CDIV(128*510*128,256),256,0,stream>>>(ws+BUF1, xt16, 128*510*128);
  k_pack_convw<<<CDIV(6*128*128,256),256,0,stream>>>((const float*)d_in[24], b16, 128, 128, 6);
  k_wmma_conv16<6><<<dim3(2,(128*505)/64),128,0,stream>>>(xt16, b16, (const float*)d_in[25],
      ws+BUF0, 510, 505);
  k_bn_stats<<<128,256,0,stream>>>(ws+BUF0, 128*505, ws+MEANV);
  k_bn_apply<<<CDIV(128*505*128,256),256,0,stream>>>(ws+BUF0, ws+MEANV,
      (const float*)d_in[28] + 128, (const float*)d_in[29] + 128, 128*505*128);
  // conv3 (KW=9): BUF0 -> BUF1
  k_cvt16<<<CDIV(128*505*128,256),256,0,stream>>>(ws+BUF0, xt16, 128*505*128);
  k_pack_convw<<<CDIV(9*128*128,256),256,0,stream>>>((const float*)d_in[26], b16, 128, 128, 9);
  k_wmma_conv16<9><<<dim3(2,(128*497)/64),128,0,stream>>>(xt16, b16, (const float*)d_in[27],
      ws+BUF1, 505, 497);
  k_bn_stats<<<128,256,0,stream>>>(ws+BUF1, 128*497, ws+MEANV);
  k_bn_apply<<<CDIV(128*497*128,256),256,0,stream>>>(ws+BUF1, ws+MEANV,
      (const float*)d_in[28] + 256, (const float*)d_in[29] + 256, 128*497*128);
  // vp is (B, L', C) already
  sem_att(ws+BUF1, (long)497*128, 128, 1, 128, 497,
          (const float*)d_in[18], (const float*)d_in[19], (const float*)d_in[20],
          out + 49152 + 128, 256);

  // ---------------- pair features + GAT ----------------
  _Float16* feat16 = (_Float16*)(ws + BUF0);          // 16,777,216 halves
  k_build_feat16<<<CDIV(32768*512,256),256,0,stream>>>(out+16384, out+49152,
      (const int*)d_in[37], feat16);
  k_cvt16_t<<<CDIV(512*128,256),256,0,stream>>>((const float*)d_in[30], b16, 512, 128); // BT 128x512
  gemm16(feat16, b16, ws+FBUF, 32768,128,512, 512,512,128, nullptr, 0);
  k_gat_elr<<<CDIV(32768*4,256),256,0,stream>>>(ws+FBUF, (const float*)d_in[31],
      (const float*)d_in[32], ws+ELR, ws+ELR+131072);
  k_fill_u32<<<CDIV(131072,256),256,0,stream>>>((unsigned*)(ws+MAXS), 0u, 131072);
  const int* knn = (const int*)d_in[41];
  k_gat_logits<<<CDIV(Etot*4,256),256,0,stream>>>(ws+ELR, ws+ELR+131072,
      knn, knn+Eknn, Eknn, Etot, ws+EBUF, (unsigned*)(ws+MAXS));
  k_fill_f32<<<CDIV(131072,256),256,0,stream>>>(ws+SUMS, 0.f, 131072);
  k_gat_exp<<<CDIV(Etot*4,256),256,0,stream>>>(ws+EBUF, (const unsigned*)(ws+MAXS),
      knn, knn+Eknn, Eknn, Etot, ws+SUMS);
  k_gat_bias<<<CDIV(32768*128,256),256,0,stream>>>(ws+OBUF, (const float*)d_in[33]);
  k_gat_agg<<<CDIV(Etot*128,256),256,0,stream>>>(ws+EBUF, ws+SUMS, ws+FBUF,
      knn, knn+Eknn, Eknn, Etot, ws+OBUF);
  k_elu<<<CDIV(32768*128,256),256,0,stream>>>(ws+OBUF, 32768*128);

  // ---------------- MLP + log_softmax ----------------
  _Float16* hcat16 = (_Float16*)(ws + BUF1);          // 4,194,304 halves
  k_cvt16<<<CDIV(32768*128,256),256,0,stream>>>(ws+OBUF, hcat16, 32768*128);
  k_cvt16_t<<<CDIV(128*32,256),256,0,stream>>>((const float*)d_in[34], b16, 128, 32); // BT 32x128
  k_wmma_gemm16_sm<<<dim3(1, 512), 32, 0, stream>>>(hcat16, b16, ws+HID,
      128, 128, 128, 32, /*elu*/2, (const int*)d_in[42]);
  k_mlp2<<<CDIV(8192,256),256,0,stream>>>(ws+HID, (const float*)d_in[35], out, 8192);
}
